// GeometryEncoder_32323923869827
// MI455X (gfx1250) — matrix operand order repaired and checked by reading, not compiled, not run on hardware
//
#include <hip/hip_runtime.h>

// ---------------------------------------------------------------------------
// GeometryEncoder for MI455X (gfx1250, wave32)
//   K1: KNN — top-k compare chain is the bottleneck (not FLOPs): 4 threads per
//       query, candidate tiles staged global->LDS with CDNA5 async copies
//       (global_load_async_to_lds_b128 / s_wait_asynccnt).
//   K2: normals via closed-form 3x3 eigh
//   K3: curvature + feature assembly -> f16 A-matrix (K padded to 32)
//   per layer: WMMA f16 GEMM (v_wmma_f32_16x16x32_f16, f32 accumulate)
//              -> column stats -> BN(+ReLU)
// Working set ~10 MB << 192 MB L2, so HBM BW is irrelevant after first touch;
// this is a latency/occupancy problem. FP16 inputs are safe: activations are
// BN-normalized (O(1)), accumulation is f32.
// ---------------------------------------------------------------------------

#define NPTS 8192
#define KNN  10
#define TPQ  4        // threads cooperating on one query
#define TILE 2048     // candidate LDS tile: 2048 pts * 12 B = 24 KB

typedef __attribute__((ext_vector_type(16))) _Float16 v16h;
typedef __attribute__((ext_vector_type(8)))  _Float16 v8h;
typedef __attribute__((ext_vector_type(8)))  float    v8f;

// ------------------------------ K1: KNN ------------------------------------
__global__ __launch_bounds__(256) void ge_knn_kernel(
    const float* __restrict__ vox, int* __restrict__ nbr) {
  __shared__ float sp[TILE * 3];        // interleaved x,y,z (flat async copy)
  __shared__ float md[256 * KNN];
  __shared__ int   mi[256 * KNN];

  const int t      = threadIdx.x;
  const int part   = t & (TPQ - 1);
  const int qlocal = t / TPQ;                 // 64 queries per block
  const int q      = blockIdx.x * (256 / TPQ) + qlocal;

  const float qx = vox[q * 3 + 0];
  const float qy = vox[q * 3 + 1];
  const float qz = vox[q * 3 + 2];

  float bd[KNN];
  int   bi[KNN];
#pragma unroll
  for (int k = 0; k < KNN; ++k) { bd[k] = 3.4e38f; bi[k] = 0; }

  const unsigned int lds_base = (unsigned int)(uintptr_t)(&sp[0]);

  for (int base = 0; base < NPTS; base += TILE) {
    __syncthreads();                          // protect prior tile reads
    {
      // CDNA5 async copy: 1536 x B128 flat memcpy (6 per thread), ASYNCcnt.
      const unsigned long long gbase =
          (unsigned long long)(uintptr_t)(vox + (size_t)base * 3);
#pragma unroll
      for (int c = 0; c < 6; ++c) {
        const int i128 = t + c * 256;
        const unsigned int       ldsa = lds_base + (unsigned int)i128 * 16u;
        const unsigned long long ga   = gbase + (unsigned long long)i128 * 16ull;
        asm volatile("global_load_async_to_lds_b128 %0, %1, off"
                     :: "v"(ldsa), "v"(ga) : "memory");
      }
      asm volatile("s_wait_asynccnt 0x0" ::: "memory");
    }
    if (base + TILE < NPTS)                    // warm GL2 for the next tile
      __builtin_prefetch(vox + (size_t)(base + TILE) * 3 + t * 8, 0, 0);
    __syncthreads();                           // publish tile to all waves

    // interleaved scan: the 4 partner lanes hit 4 distinct LDS banks,
    // identical addresses across query groups broadcast.
    for (int jj = part; jj < TILE; jj += TPQ) {
      const int j = base + jj;
      const float dx = qx - sp[jj * 3 + 0];
      const float dy = qy - sp[jj * 3 + 1];
      const float dz = qz - sp[jj * 3 + 2];
      const float d2 = dx * dx + dy * dy + dz * dz;
      if (j == q) continue;                   // exclude self (ref drops idx 0)
      if (d2 < bd[KNN - 1]) {
        int p = KNN - 1;
        while (p > 0 && bd[p - 1] > d2) { bd[p] = bd[p - 1]; bi[p] = bi[p - 1]; --p; }
        bd[p] = d2; bi[p] = j;
      }
    }
  }

  // publish local sorted lists, then part==0 merges TPQ lists of KNN
#pragma unroll
  for (int k = 0; k < KNN; ++k) { md[t * KNN + k] = bd[k]; mi[t * KNN + k] = bi[k]; }
  __syncthreads();
  if (part == 0) {
    int p[TPQ];
#pragma unroll
    for (int s = 0; s < TPQ; ++s) p[s] = 0;
    for (int k = 0; k < KNN; ++k) {
      float best = 3.5e38f; int bsel = 0;
#pragma unroll
      for (int s = 0; s < TPQ; ++s) {
        if (p[s] < KNN) {
          const float v = md[(t + s) * KNN + p[s]];
          if (v < best) { best = v; bsel = s; }
        }
      }
      nbr[q * KNN + k] = mi[(t + bsel) * KNN + p[bsel]];
      p[bsel]++;
    }
  }
}

// ------------------------- K2: normals (3x3 eigh) --------------------------
__device__ inline void ge_cross(float ax, float ay, float az,
                                float bx, float by, float bz,
                                float& cx, float& cy, float& cz) {
  cx = ay * bz - az * by;
  cy = az * bx - ax * bz;
  cz = ax * by - ay * bx;
}

__global__ __launch_bounds__(256) void ge_normals_kernel(
    const float* __restrict__ vox, const int* __restrict__ nbr,
    float* __restrict__ normals) {
  const int i = blockIdx.x * 256 + threadIdx.x;
  const float px = vox[i * 3 + 0];
  const float py = vox[i * 3 + 1];
  const float pz = vox[i * 3 + 2];

  float c00 = 0.f, c01 = 0.f, c02 = 0.f, c11 = 0.f, c12 = 0.f, c22 = 0.f;
#pragma unroll
  for (int k = 0; k < KNN; ++k) {
    const int j = nbr[i * KNN + k];
    const float lx = vox[j * 3 + 0] - px;
    const float ly = vox[j * 3 + 1] - py;
    const float lz = vox[j * 3 + 2] - pz;
    c00 += lx * lx; c01 += lx * ly; c02 += lx * lz;
    c11 += ly * ly; c12 += ly * lz; c22 += lz * lz;
  }
  const float inv = 1.0f / (KNN - 1);
  c00 *= inv; c01 *= inv; c02 *= inv; c11 *= inv; c12 *= inv; c22 *= inv;

  // smallest eigenvalue (trigonometric closed form)
  const float p1 = c01 * c01 + c02 * c02 + c12 * c12;
  const float qm = (c00 + c11 + c22) * (1.0f / 3.0f);
  const float d0 = c00 - qm, d1 = c11 - qm, d2 = c22 - qm;
  const float p2 = d0 * d0 + d1 * d1 + d2 * d2 + 2.f * p1;
  const float p  = sqrtf(p2 * (1.0f / 6.0f));
  float lmin = qm;
  if (p > 1e-20f) {
    const float ip = 1.0f / p;
    const float b00 = d0 * ip, b01 = c01 * ip, b02 = c02 * ip;
    const float b11 = d1 * ip, b12 = c12 * ip, b22 = d2 * ip;
    float detB = b00 * (b11 * b22 - b12 * b12)
               - b01 * (b01 * b22 - b12 * b02)
               + b02 * (b01 * b12 - b11 * b02);
    float r = 0.5f * detB;
    r = fminf(1.f, fmaxf(-1.f, r));
    const float phi = acosf(r) * (1.0f / 3.0f);
    lmin = qm + 2.f * p * cosf(phi + 2.0943951023931953f); // +2pi/3 -> smallest
  }

  // eigenvector of (A - lmin I): best cross product of rows
  const float r0x = c00 - lmin, r0y = c01,        r0z = c02;
  const float r1x = c01,        r1y = c11 - lmin, r1z = c12;
  const float r2x = c02,        r2y = c12,        r2z = c22 - lmin;
  float ax, ay, az, bx, by, bz, cx, cy, cz;
  ge_cross(r0x, r0y, r0z, r1x, r1y, r1z, ax, ay, az);
  ge_cross(r0x, r0y, r0z, r2x, r2y, r2z, bx, by, bz);
  ge_cross(r1x, r1y, r1z, r2x, r2y, r2z, cx, cy, cz);
  const float na = ax * ax + ay * ay + az * az;
  const float nb = bx * bx + by * by + bz * bz;
  const float nc = cx * cx + cy * cy + cz * cz;
  float vx = ax, vy = ay, vz = az, vn = na;
  if (nb > vn) { vx = bx; vy = by; vz = bz; vn = nb; }
  if (nc > vn) { vx = cx; vy = cy; vz = cz; vn = nc; }
  if (vn < 1e-30f) { vx = 1.f; vy = 0.f; vz = 0.f; vn = 1.f; }
  const float rn = 1.0f / fmaxf(sqrtf(vn), 1e-12f);
  normals[i * 3 + 0] = vx * rn;
  normals[i * 3 + 1] = vy * rn;
  normals[i * 3 + 2] = vz * rn;
}

// ----------------- K3: curvature + features -> f16 A-matrix ----------------
// featsA: row-major NPTS x 32 f16, cols 0..7 = [x y z nx ny nz curv dist],
// cols 8..31 zero (K padded for 16x16x32 WMMA).
__global__ __launch_bounds__(256) void ge_feats_kernel(
    const float* __restrict__ vox, const float* __restrict__ normals,
    const int* __restrict__ nbr, _Float16* __restrict__ featsA) {
  const int i = blockIdx.x * 256 + threadIdx.x;
  const float x = vox[i * 3 + 0], y = vox[i * 3 + 1], z = vox[i * 3 + 2];
  const float nx = normals[i * 3 + 0], ny = normals[i * 3 + 1], nz = normals[i * 3 + 2];
  float curv = 0.f;
#pragma unroll
  for (int k = 0; k < KNN; ++k) {
    const int j = nbr[i * KNN + k];
    const float dx = normals[j * 3 + 0] - nx;
    const float dy = normals[j * 3 + 1] - ny;
    const float dz = normals[j * 3 + 2] - nz;
    curv += sqrtf(dx * dx + dy * dy + dz * dz);
  }
  curv *= (1.0f / KNN);
  const float dist = sqrtf(x * x + y * y + z * z);

  float f[8] = {x, y, z, nx, ny, nz, curv, dist};
  _Float16* row = featsA + (size_t)i * 32;
#pragma unroll
  for (int c = 0; c < 8; ++c)  row[c] = (_Float16)f[c];
#pragma unroll
  for (int c = 8; c < 32; ++c) row[c] = (_Float16)0.f;
}

// ------------- weight f32 -> f16, TRANSPOSED + K-padded --------------------
// Wt[n*Kpad + k] so a B-fragment lane reads two runs of 8 K-contiguous halves (2x b128).
__global__ __launch_bounds__(256) void ge_prep_w_kernel(
    const float* __restrict__ W, _Float16* __restrict__ Wt,
    int Kin, int Kpad, int Ncols) {
  const int idx = blockIdx.x * 256 + threadIdx.x;
  if (idx >= Kpad * Ncols) return;
  const int n = idx / Kpad;
  const int k = idx - n * Kpad;
  Wt[idx] = (k < Kin) ? (_Float16)W[k * Ncols + n] : (_Float16)0.f;
}

// -------------------- WMMA GEMM: Z = A(f16) @ W(f16) (+bias) ---------------
// 4 waves per block, each wave owns one 16x16 output tile. K loop fully
// unrolled (KPAD is a template param) -> back-to-back v_wmma for KPAD=64.
template <int KPAD, bool HAS_BIAS>
__global__ __launch_bounds__(128) void ge_gemm_f16_kernel(
    const _Float16* __restrict__ A, const _Float16* __restrict__ Wt,
    const float* __restrict__ bias, float* __restrict__ Z, int Ncols) {
  const int lane  = threadIdx.x & 31;
  const int wave  = threadIdx.x >> 5;
  const int mtile = (blockIdx.x * 4 + wave) * 16;
  const int ntile = blockIdx.y * 16;
  const int l15   = lane & 15;
  const bool hi   = (lane >= 16);

  v8f acc = {};
#pragma unroll
  for (int kt = 0; kt < KPAD; kt += 32) {
    // A fragment (16x32 f16): lane<16 -> K {0..7,16..23}, lane>=16 -> {8..15,24..31}
    const int arow = mtile + l15;
    const int klo  = kt + (hi ? 8 : 0);
    const v8h alo = *reinterpret_cast<const v8h*>(A + (size_t)arow * KPAD + klo);
    const v8h ahi = *reinterpret_cast<const v8h*>(A + (size_t)arow * KPAD + klo + 16);
    v16h a;
#pragma unroll
    for (int e = 0; e < 8; ++e) { a[e] = alo[e]; a[e + 8] = ahi[e]; }

    // B fragment (32x16 f16): lane<16 -> K {0..7,16..23} of col N, lane>=16 -> {8..15,24..31}.
    // Wt is N-major, so each 8-half K run is contiguous: two 16-byte loads, 16 halves apart.
    const int n    = ntile + l15;
    const int koff = kt + (hi ? 8 : 0);
    const _Float16* wp = Wt + (size_t)n * KPAD + koff;
    const v8h blo = *reinterpret_cast<const v8h*>(wp);
    const v8h bhi = *reinterpret_cast<const v8h*>(wp + 16);
    v16h b;
#pragma unroll
    for (int e = 0; e < 8; ++e) { b[e] = blo[e]; b[e + 8] = bhi[e]; }

    acc = __builtin_amdgcn_wmma_f32_16x16x32_f16(
        /*neg_a=*/false, a, /*neg_b=*/false, b,
        /*c_mod=*/(short)0, acc, /*reuse_a=*/false, /*reuse_b=*/false);
  }

  // C/D layout: VGPR v: lanes 0-15 -> (M=v, N=lane); lanes 16-31 -> (M=v+8)
#pragma unroll
  for (int v = 0; v < 8; ++v) {
    const int r = mtile + v + (hi ? 8 : 0);
    const int c = ntile + l15;
    float val = acc[v];
    if (HAS_BIAS) val += bias[c];
    Z[(size_t)r * Ncols + c] = val;
  }
}

// ----------------------- per-column mean / variance ------------------------
__global__ __launch_bounds__(256) void ge_colstats_kernel(
    const float* __restrict__ Z, int Ncols,
    float* __restrict__ mean, float* __restrict__ var) {
  __shared__ float ss[256], ss2[256];
  const int c = blockIdx.x;
  float s = 0.f, s2 = 0.f;
  for (int r = threadIdx.x; r < NPTS; r += 256) {
    const float v = Z[(size_t)r * Ncols + c];
    s += v; s2 += v * v;
  }
  ss[threadIdx.x] = s; ss2[threadIdx.x] = s2;
  __syncthreads();
  for (int off = 128; off > 0; off >>= 1) {
    if (threadIdx.x < off) {
      ss[threadIdx.x]  += ss[threadIdx.x + off];
      ss2[threadIdx.x] += ss2[threadIdx.x + off];
    }
    __syncthreads();
  }
  if (threadIdx.x == 0) {
    const float m = ss[0] * (1.0f / NPTS);
    mean[c] = m;
    var[c]  = ss2[0] * (1.0f / NPTS) - m * m;
  }
}

// -------------------- BN (+ReLU) -> f16 activations ------------------------
__global__ __launch_bounds__(256) void ge_bn_relu_f16_kernel(
    const float* __restrict__ Z, const float* __restrict__ mean,
    const float* __restrict__ var, const float* __restrict__ gamma,
    const float* __restrict__ beta, _Float16* __restrict__ H, int Ncols) {
  const int idx = blockIdx.x * 256 + threadIdx.x;
  if (idx >= NPTS * Ncols) return;
  const int c = idx & (Ncols - 1);   // Ncols is a power of two
  float v = gamma[c] * (Z[idx] - mean[c]) * rsqrtf(var[c] + 1e-5f) + beta[c];
  v = fmaxf(v, 0.f);
  H[idx] = (_Float16)v;
}

// ----------------------- final BN -> f32 output ----------------------------
__global__ __launch_bounds__(256) void ge_bn_out_kernel(
    const float* __restrict__ Z, const float* __restrict__ mean,
    const float* __restrict__ var, const float* __restrict__ gamma,
    const float* __restrict__ beta, float* __restrict__ out, int Ncols) {
  const int idx = blockIdx.x * 256 + threadIdx.x;
  if (idx >= NPTS * Ncols) return;
  const int c = idx & (Ncols - 1);
  out[idx] = gamma[c] * (Z[idx] - mean[c]) * rsqrtf(var[c] + 1e-5f) + beta[c];
}

// ---------------------------------------------------------------------------
extern "C" void kernel_launch(void* const* d_in, const int* in_sizes, int n_in,
                              void* d_out, int out_size, void* d_ws, size_t ws_size,
                              hipStream_t stream) {
  const float* voxels = (const float*)d_in[0];
  const float* W0     = (const float*)d_in[1];
  const float* g0     = (const float*)d_in[2];
  const float* b0     = (const float*)d_in[3];
  const float* W1     = (const float*)d_in[4];
  const float* g1     = (const float*)d_in[5];
  const float* b1     = (const float*)d_in[6];
  const float* W2     = (const float*)d_in[7];
  const float* bias2  = (const float*)d_in[8];
  const float* g2     = (const float*)d_in[9];
  const float* b2     = (const float*)d_in[10];
  float* out = (float*)d_out;

  // workspace layout (512B-aligned slabs)
  char* ws = (char*)d_ws;
  size_t off = 0;
  auto alloc = [&](size_t bytes) {
    void* p = ws + off;
    off += (bytes + 511) & ~(size_t)511;
    return p;
  };
  int*       nbr     = (int*)      alloc((size_t)NPTS * KNN * 4);
  float*     normals = (float*)    alloc((size_t)NPTS * 3 * 4);
  _Float16*  featsA  = (_Float16*) alloc((size_t)NPTS * 32 * 2);
  _Float16*  W0t     = (_Float16*) alloc((size_t)32 * 32 * 2);
  _Float16*  W1t     = (_Float16*) alloc((size_t)64 * 32 * 2);
  _Float16*  W2t     = (_Float16*) alloc((size_t)128 * 64 * 2);
  float*     z0      = (float*)    alloc((size_t)NPTS * 32 * 4);
  _Float16*  h0      = (_Float16*) alloc((size_t)NPTS * 32 * 2);
  float*     z1      = (float*)    alloc((size_t)NPTS * 64 * 4);
  _Float16*  h1      = (_Float16*) alloc((size_t)NPTS * 64 * 2);
  float*     z2      = (float*)    alloc((size_t)NPTS * 128 * 4);
  float*     mean0   = (float*)    alloc(32 * 4);
  float*     var0    = (float*)    alloc(32 * 4);
  float*     mean1   = (float*)    alloc(64 * 4);
  float*     var1    = (float*)    alloc(64 * 4);
  float*     mean2   = (float*)    alloc(128 * 4);
  float*     var2    = (float*)    alloc(128 * 4);
  (void)ws_size; (void)in_sizes; (void)n_in; (void)out_size;

  // weight prep (f32 -> f16 transposed, K padded)
  ge_prep_w_kernel<<<(32 * 32 + 255) / 256, 256, 0, stream>>>(W0, W0t, 8, 32, 32);
  ge_prep_w_kernel<<<(64 * 32 + 255) / 256, 256, 0, stream>>>(W1, W1t, 32, 32, 64);
  ge_prep_w_kernel<<<(128 * 64 + 255) / 256, 256, 0, stream>>>(W2, W2t, 64, 64, 128);

  // geometry pipeline
  ge_knn_kernel<<<NPTS / (256 / TPQ), 256, 0, stream>>>(voxels, nbr);
  ge_normals_kernel<<<NPTS / 256, 256, 0, stream>>>(voxels, nbr, normals);
  ge_feats_kernel<<<NPTS / 256, 256, 0, stream>>>(voxels, normals, nbr, featsA);

  // layer 0: 8(pad32) -> 32
  ge_gemm_f16_kernel<32, false><<<dim3(NPTS / 64, 32 / 16), 128, 0, stream>>>(
      featsA, W0t, nullptr, z0, 32);
  ge_colstats_kernel<<<32, 256, 0, stream>>>(z0, 32, mean0, var0);
  ge_bn_relu_f16_kernel<<<(NPTS * 32 + 255) / 256, 256, 0, stream>>>(
      z0, mean0, var0, g0, b0, h0, 32);

  // layer 1: 32 -> 64
  ge_gemm_f16_kernel<32, false><<<dim3(NPTS / 64, 64 / 16), 128, 0, stream>>>(
      h0, W1t, nullptr, z1, 64);
  ge_colstats_kernel<<<64, 256, 0, stream>>>(z1, 64, mean1, var1);
  ge_bn_relu_f16_kernel<<<(NPTS * 64 + 255) / 256, 256, 0, stream>>>(
      z1, mean1, var1, g1, b1, h1, 64);

  // layer 2: 64 -> 128 (+bias), BN (no relu) -> d_out
  ge_gemm_f16_kernel<64, true><<<dim3(NPTS / 64, 128 / 16), 128, 0, stream>>>(
      h1, W2t, bias2, z2, 128);
  ge_colstats_kernel<<<128, 256, 0, stream>>>(z2, 128, mean2, var2);
  ge_bn_out_kernel<<<(NPTS * 128 + 255) / 256, 256, 0, stream>>>(
      z2, mean2, var2, g2, b2, out, 128);
}
